// CoxLossBreslow_54159537602912
// MI455X (gfx1250) — matrix operand order repaired and checked by reading, not compiled, not run on hardware
//
#include <hip/hip_runtime.h>
#include <hip/hip_bf16.h>
#include <math.h>

typedef __attribute__((ext_vector_type(16))) _Float16 v16h;
typedef __attribute__((ext_vector_type(8)))  float    v8f;
typedef __attribute__((ext_vector_type(4)))  float    v4f;

// ---------------------------------------------------------------------------
// Kernel 1: extract event times, compute exp(est) split into f16 hi/lo parts.
// Pads [N, Nk) with zeros so padded K-columns contribute nothing.
// ---------------------------------------------------------------------------
__global__ void cox_prep(const float* __restrict__ est,
                         const float* __restrict__ target,
                         float* __restrict__ t,
                         _Float16* __restrict__ ehi,
                         _Float16* __restrict__ elo,
                         int N, int Nk)
{
    int i = blockIdx.x * blockDim.x + threadIdx.x;
    if (i >= Nk) return;
    if (i < N) {
        t[i] = target[2 * i];                 // event_time
        float e = expf(est[i]);
        _Float16 h = (_Float16)e;
        ehi[i] = h;
        elo[i] = (_Float16)(e - (float)h);    // residual -> ~f32 precision total
    } else {
        t[i]   = 0.0f;
        ehi[i] = (_Float16)0.0f;
        elo[i] = (_Float16)0.0f;
    }
}

// ---------------------------------------------------------------------------
// Kernel 2: denom[i] = sum_j [t_i <= t_j] * exp(est_j) via WMMA f16->f32.
// One wave per 16-row tile. A = exact {0,1} mask (16x32 f16). B packs the
// hi-split exp values into columns 0-7 and the lo-split into columns 8-15,
// so ONE WMMA per 32-wide K chunk accumulates both partial sums; a final
// shfl_xor(8) combines hi+lo columns. Two accumulators (even/odd chunks)
// keep consecutive WMMAs independent so they pipeline.
// ---------------------------------------------------------------------------
__global__ void __launch_bounds__(32)
cox_denom(const float* __restrict__ t,
          const _Float16* __restrict__ ehi,
          const _Float16* __restrict__ elo,
          float* __restrict__ denom, int Nk)
{
    const int lane = threadIdx.x;          // 32-thread block == one wave32
    const int row0 = blockIdx.x * 16;
    const int m    = lane & 15;            // A row M / C,D column N
    const int half = lane >> 4;            // 0: lanes 0-15, 1: lanes 16-31
    const int kA   = half * 16;            // A K-offset for this lane group
    const int kB   = half * 16;            // B K-offset for this lane group
    const float trow = t[row0 + m];
    // B columns 0-7 carry exp-hi, columns 8-15 carry exp-lo.
    const _Float16* __restrict__ esrc = (m < 8) ? ehi : elo;

    v8f acc0 = {0.f, 0.f, 0.f, 0.f, 0.f, 0.f, 0.f, 0.f};
    v8f acc1 = {0.f, 0.f, 0.f, 0.f, 0.f, 0.f, 0.f, 0.f};

    const _Float16 one  = (_Float16)1.0f;
    const _Float16 zero = (_Float16)0.0f;

    for (int kc = 0; kc < Nk; kc += 64) {
#pragma unroll
        for (int u = 0; u < 2; ++u) {
            const int kb = kc + u * 32;
            // Times for this lane's 16 K-slots:
            //   elements 0-15 -> K = kb+kA+ 0..15
            const v4f* tv = (const v4f*)(t + kb + kA);
            v4f t0 = tv[0];
            v4f t1 = tv[1];
            v4f t2 = tv[2];
            v4f t3 = tv[3];

            v16h a;
#pragma unroll
            for (int e = 0; e < 4; ++e) {
                a[e]      = (trow <= t0[e]) ? one : zero;
                a[e + 4]  = (trow <= t1[e]) ? one : zero;
                a[e + 8]  = (trow <= t2[e]) ? one : zero;
                a[e + 12] = (trow <= t3[e]) ? one : zero;
            }

            // B (32x16 f16): lane supplies column n=m; elements e -> K=kB+e.
            v16h b = *(const v16h*)(esrc + kb + kB);

            if (u == 0)
                acc0 = __builtin_amdgcn_wmma_f32_16x16x32_f16(
                    false, a, false, b, (short)0, acc0, false, false);
            else
                acc1 = __builtin_amdgcn_wmma_f32_16x16x32_f16(
                    false, a, false, b, (short)0, acc1, false, false);
        }
    }

    // Combine even/odd chunk accumulators, then hi(cols 0-7)+lo(cols 8-15)
    // via lane pairing n <-> n^8 (stays within the same M half-group).
    float r[8];
#pragma unroll
    for (int v = 0; v < 8; ++v) {
        float x = acc0[v] + acc1[v];
        x += __shfl_xor(x, 8, 32);
        r[v] = x;
    }

    // Lanes with n==0 (lanes 0 and 16) hold rows M = v + 8*half in slot v.
    if (m == 0) {
        const int mbase = half * 8;
#pragma unroll
        for (int v = 0; v < 8; ++v)
            denom[row0 + mbase + v] = r[v];
    }
}

// ---------------------------------------------------------------------------
// Kernel 3: loss = sum_i ev_i * (log(denom_i) - est_i) / max(sum ev, 1)
// ---------------------------------------------------------------------------
__global__ void cox_finalize(const float* __restrict__ est,
                             const float* __restrict__ target,
                             const float* __restrict__ denom,
                             float* __restrict__ out, int N)
{
    __shared__ float s_num[256];
    __shared__ float s_cnt[256];
    const int tid = threadIdx.x;
    float num = 0.0f, cnt = 0.0f;
    for (int i = tid; i < N; i += 256) {
        float ev = (target[2 * i + 1] != 0.0f) ? 1.0f : 0.0f;
        num += ev * (logf(denom[i]) - est[i]);
        cnt += ev;
    }
    s_num[tid] = num;
    s_cnt[tid] = cnt;
    __syncthreads();
    for (int s = 128; s > 0; s >>= 1) {
        if (tid < s) {
            s_num[tid] += s_num[tid + s];
            s_cnt[tid] += s_cnt[tid + s];
        }
        __syncthreads();
    }
    if (tid == 0) out[0] = s_num[0] / fmaxf(s_cnt[0], 1.0f);
}

// ---------------------------------------------------------------------------
extern "C" void kernel_launch(void* const* d_in, const int* in_sizes, int n_in,
                              void* d_out, int out_size, void* d_ws, size_t ws_size,
                              hipStream_t stream)
{
    const float* est    = (const float*)d_in[0];
    const float* target = (const float*)d_in[1];
    const int N  = in_sizes[0];
    const int Nk = (N + 63) & ~63;         // pad K to 2 WMMA chunks (64)

    // Workspace layout (Nk*12 bytes total; ~192 KB for N=16384):
    //   [0,        Nk*4)  : t     (f32 event times)
    //   [Nk*4,     Nk*6)  : ehi   (f16 exp hi)
    //   [Nk*6,     Nk*8)  : elo   (f16 exp lo)
    //   [Nk*8,     Nk*12) : denom (f32)
    char* ws = (char*)d_ws;
    float*    t     = (float*)ws;
    _Float16* ehi   = (_Float16*)(ws + (size_t)Nk * 4);
    _Float16* elo   = (_Float16*)(ws + (size_t)Nk * 6);
    float*    denom = (float*)(ws + (size_t)Nk * 8);

    cox_prep<<<(Nk + 255) / 256, 256, 0, stream>>>(est, target, t, ehi, elo, N, Nk);
    cox_denom<<<Nk / 16, 32, 0, stream>>>(t, ehi, elo, denom, Nk);
    cox_finalize<<<1, 256, 0, stream>>>(est, target, denom, (float*)d_out, N);
}
